// NLBlock_60378650247803
// MI455X (gfx1250) — compile-verified, matrix operand order checked by reading, not run on hardware
//
#include <hip/hip_runtime.h>

// ---------------------------------------------------------------------------
// Non-local block (video self-attention), MI455X / gfx1250, wave32 + WMMA.
// All GEMMs use v_wmma_f32_16x16x32_bf16 (bf16 in, fp32 accumulate).
// Attention is fused flash-style: the 4x12544x1568 score matrix (315 MB fp32)
// is never materialized in HBM -- on a 23.3 TB/s part that is the dominant
// cost of the naive schedule. BN is folded into bf16 weights once.
// Tile staging uses GLOBAL_LOAD_ASYNC_TO_LDS_B128 (ASYNCcnt path).
// ---------------------------------------------------------------------------

typedef unsigned short u16;
typedef unsigned int   u32;

typedef __attribute__((ext_vector_type(16))) __bf16 v16bf;
typedef __attribute__((ext_vector_type(8)))  float  v8f;
typedef __attribute__((ext_vector_type(4)))  int    i32x4;

#define B_N   4
#define C_IN  256
#define C_I   128
#define N_SP  12544   // 16*28*28
#define M_SP  1568    // 8*14*14
#define EPSV  1e-5f

#if defined(__has_builtin)
#  if __has_builtin(__builtin_amdgcn_global_load_async_to_lds_b128)
#    define HAVE_ASYNC_LDS 1
#  endif
#endif
#ifndef HAVE_ASYNC_LDS
#  define HAVE_ASYNC_LDS 0
#endif

__device__ __forceinline__ u16 f2bf(float f) {
    u32 u = __float_as_uint(f);
    u32 r = u + 0x7FFFu + ((u >> 16) & 1u);   // round-to-nearest-even
    return (u16)(r >> 16);
}
__device__ __forceinline__ float bf2f(u16 h) {
    return __uint_as_float(((u32)h) << 16);
}

union FragBF { v16bf v; uint4 q[2]; u16 u[16]; };

// Load a 16x32 bf16 WMMA operand fragment from a row-major LDS tile.
// ISA layout (CDNA5 ISA): lanes 0-15 row=lane, K 0-7 & 16-23;
// lanes 16-31 same rows, K 8-15 & 24-31 -> two contiguous 16B chunks/lane.
__device__ __forceinline__ v16bf load_frag16(const u16* base, int ld, int r0, int k0) {
    const int lid = threadIdx.x & 31;
    const int row = lid & 15;
    const int hi8 = (lid >> 4) << 3;
    const u16* p = base + (r0 + row) * ld + k0 + hi8;
    FragBF f;
    f.q[0] = *(const uint4*)(p);
    f.q[1] = *(const uint4*)(p + 16);
    return f.v;
}

__device__ __forceinline__ v8f zero_v8() {
    v8f z;
#pragma unroll
    for (int i = 0; i < 8; ++i) z[i] = 0.f;
    return z;
}

__device__ __forceinline__ v8f wmma_bf16(v16bf a, v16bf b, v8f c) {
    return __builtin_amdgcn_wmma_f32_16x16x32_bf16(false, a, false, b,
                                                   (short)0, c, false, false);
}

// 16B global->LDS copy: async path (ASYNCcnt) when available.
__device__ __forceinline__ void cp_b128(const u16* g, u16* l) {
#if HAVE_ASYNC_LDS
    __builtin_amdgcn_global_load_async_to_lds_b128(
        (__attribute__((address_space(1))) i32x4*)(void*)g,
        (__attribute__((address_space(3))) i32x4*)(void*)l, 0, 0);
#else
    *(uint4*)l = *(const uint4*)g;
#endif
}
__device__ __forceinline__ void cp_wait() {
#if HAVE_ASYNC_LDS
    asm volatile("s_wait_asynccnt 0x0" ::: "memory");
#endif
}

// ---------------------------------------------------------------------------
// Kernel 1: fold BN into weights, convert to bf16.
// Wcat[384][256] = rows 0-127 theta, 128-255 phi, 256-383 g (BN1 folded).
// Ww[256][128]   = output conv (BN2 folded).
// ---------------------------------------------------------------------------
__global__ void k_fold(const float* w_theta, const float* b_theta,
                       const float* w_phi,   const float* b_phi,
                       const float* w_g,     const float* b_g,
                       const float* g1, const float* be1,
                       const float* mu1, const float* va1,
                       const float* w_W, const float* b_W,
                       const float* g2, const float* be2,
                       const float* mu2, const float* va2,
                       u16* Wcat, float* bias1, u16* Ww, float* biasW) {
    const int bid = blockIdx.x, tid = threadIdx.x;
    if (bid < 384) {
        const int grp = bid >> 7, c = bid & 127;
        const float* w  = (grp == 0) ? w_theta : (grp == 1) ? w_phi : w_g;
        const float* bb = (grp == 0) ? b_theta : (grp == 1) ? b_phi : b_g;
        const float inv = g1[c] * rsqrtf(va1[c] + EPSV);
        Wcat[bid * 256 + tid] = f2bf(w[c * 256 + tid] * inv);
        if (tid == 0) bias1[bid] = bb[c] * inv + be1[c] - mu1[c] * inv;
    } else {
        const int co = bid - 384;
        const float inv = g2[co] * rsqrtf(va2[co] + EPSV);
        if (tid < 128) Ww[co * 128 + tid] = f2bf(w_W[co * 128 + tid] * inv);
        if (tid == 0) biasW[co] = b_W[co] * inv + be2[co] - mu2[co] * inv;
    }
}

// ---------------------------------------------------------------------------
// Kernel 2: x (fp32) -> bf16.
// ---------------------------------------------------------------------------
__global__ void k_cvt(const float* __restrict__ x, u16* __restrict__ xb, int n) {
    const int i = blockIdx.x * 256 + threadIdx.x;
    if (i < n) xb[i] = f2bf(x[i]);
}

// ---------------------------------------------------------------------------
// Kernel 3: projections. P[b][r][n] = relu(Wcat x + bias1), r in [0,384).
// Tile: 128 rows x 128 cols per block; K=256 in 8 steps of 32.
// 8 waves; wave w owns rows [16w,16w+16).
// ---------------------------------------------------------------------------
__global__ void __launch_bounds__(256) k_proj(const u16* __restrict__ Wcat,
                                              const float* __restrict__ bias1,
                                              const u16* __restrict__ Xbf,
                                              u16* __restrict__ P) {
    __shared__ __align__(16) u16 As[128 * 32];
    __shared__ __align__(16) u16 Bs[128 * 32];   // [n][c] (transposed on load)
    const int n0  = blockIdx.x * 128;
    const int r0  = blockIdx.y * 128;
    const int b   = blockIdx.z;
    const int tid = threadIdx.x;
    const int w = tid >> 5, lid = tid & 31;
    const int col = lid & 15, hi8 = (lid >> 4) << 3;

    v8f acc[8];
#pragma unroll
    for (int t = 0; t < 8; ++t) acc[t] = zero_v8();

    for (int ks = 0; ks < 8; ++ks) {
        if (ks + 1 < 8) {   // prefetch next k-step tiles (global_prefetch_b8)
            __builtin_prefetch(Wcat + (r0 + (tid >> 1)) * 256 + (ks + 1) * 32, 0, 3);
            __builtin_prefetch(Xbf + ((size_t)(b * 256 + (ks + 1) * 32 + (tid >> 3))) * N_SP + n0, 0, 3);
        }
        __syncthreads();
        {   // A tile: Wcat rows (row-major over k) -> direct async copy
            const int row = tid >> 1, seg = (tid & 1) * 16;
            const u16* src = Wcat + (r0 + row) * 256 + ks * 32 + seg;
            u16* dst = As + row * 32 + seg;
            cp_b128(src, dst);
            cp_b128(src + 8, dst + 8);
        }
        {   // B tile: Xbf[c][n] -> Bs[n][c] (transpose; global reads coalesced)
            const int c = tid >> 3, nseg = (tid & 7) * 16;
            const u16* src = Xbf + ((size_t)(b * 256 + ks * 32 + c)) * N_SP + n0 + nseg;
            FragBF t0;
            t0.q[0] = ((const uint4*)src)[0];
            t0.q[1] = ((const uint4*)src)[1];
#pragma unroll
            for (int j = 0; j < 16; ++j) Bs[(nseg + j) * 32 + c] = t0.u[j];
        }
        cp_wait();
        __syncthreads();
        const v16bf a = load_frag16(As, 32, w * 16, 0);
        v16bf bfr[8];
#pragma unroll
        for (int t = 0; t < 8; ++t) bfr[t] = load_frag16(Bs, 32, t * 16, 0);
#pragma unroll
        for (int t = 0; t < 8; ++t) acc[t] = wmma_bf16(a, bfr[t], acc[t]);
    }
#pragma unroll
    for (int t = 0; t < 8; ++t) {
#pragma unroll
        for (int r = 0; r < 8; ++r) {
            const int row = r0 + w * 16 + r + hi8;
            float v = acc[t][r] + bias1[row];
            v = fmaxf(v, 0.f);
            P[((size_t)(b * 384 + row)) * N_SP + n0 + t * 16 + col] = f2bf(v);
        }
    }
}

// ---------------------------------------------------------------------------
// Kernel 4: 2x2x2 maxpool of phi and g.
// KpT[b][m][c] (m-major -> QK^T B-fragments load straight from LDS),
// Gp [b][c][m] (c-major -> PV     B-fragments load straight from LDS).
// ---------------------------------------------------------------------------
__global__ void k_pool(const u16* __restrict__ P,
                       u16* __restrict__ KpT, u16* __restrict__ Gp) {
    const int i = blockIdx.x * 256 + threadIdx.x;
    if (i >= B_N * C_I * M_SP) return;
    const int b   = i / (C_I * M_SP);
    const int rem = i % (C_I * M_SP);
    const int c   = rem / M_SP;
    const int m   = rem % M_SP;
    const int tp = m / 196, hp = (m % 196) / 14, wp = m % 14;
    const u16* Pphi = P + ((size_t)(b * 384 + 128 + c)) * N_SP;
    const u16* Pg   = P + ((size_t)(b * 384 + 256 + c)) * N_SP;
    float mphi = -3.0e38f, mg = -3.0e38f;
#pragma unroll
    for (int dt = 0; dt < 2; ++dt)
#pragma unroll
        for (int dh = 0; dh < 2; ++dh)
#pragma unroll
            for (int dw = 0; dw < 2; ++dw) {
                const int n = (2 * tp + dt) * 784 + (2 * hp + dh) * 28 + (2 * wp + dw);
                mphi = fmaxf(mphi, bf2f(Pphi[n]));
                mg   = fmaxf(mg,   bf2f(Pg[n]));
            }
    KpT[((size_t)(b * M_SP + m)) * 128 + c] = f2bf(mphi);
    Gp [((size_t)(b * 128 + c)) * M_SP + m] = f2bf(mg);
}

// ---------------------------------------------------------------------------
// Kernel 5: fused flash attention.
// 128 queries per block; m in 13 tiles of 128 (last masked to 1568).
// Q fragments kept resident in registers; P=exp(S-max) restaged via per-wave
// LDS strip to form PV A-fragments. Y[b][n][c] bf16 (n-major for kernel 6).
// LDS: 32KB (Q-stage/K) + 32KB (G) + 32KB (P strips) = 96KB.
// ---------------------------------------------------------------------------
__global__ void __launch_bounds__(256) k_attn(const u16* __restrict__ P,
                                              const u16* __restrict__ KpT,
                                              const u16* __restrict__ Gp,
                                              u16* __restrict__ Y) {
    __shared__ __align__(16) u16 Ks[128 * 128];      // Q staging, then K tiles [m][c]
    __shared__ __align__(16) u16 Gs[128 * 128];      // G tiles [c][m]
    __shared__ __align__(16) u16 Ps[8 * 16 * 128];   // per-wave exp(S) strips [q][m]
    const int n0  = blockIdx.x * 128;
    const int b   = blockIdx.y;
    const int tid = threadIdx.x;
    const int w = tid >> 5, lid = tid & 31;
    const int col = lid & 15, hi8 = (lid >> 4) << 3;

    // Stage Q (theta transposed): Ks[q][c] = P[b][c][n0+q]
    for (int i = 0; i < 64; ++i) {
        const int idx = tid * 64 + i;
        const int q = idx & 127, c = idx >> 7;
        Ks[q * 128 + c] = P[((size_t)(b * 384 + c)) * N_SP + n0 + q];
    }
    __syncthreads();
    v16bf qf[4];
#pragma unroll
    for (int kk = 0; kk < 4; ++kk) qf[kk] = load_frag16(Ks, 128, w * 16, kk * 32);

    v8f O[8];
#pragma unroll
    for (int t = 0; t < 8; ++t) O[t] = zero_v8();
    float rmax[8], rsum[8];
#pragma unroll
    for (int r = 0; r < 8; ++r) { rmax[r] = -3.0e38f; rsum[r] = 0.f; }

    u16* Pw = Ps + w * (16 * 128);

    for (int it = 0; it < 13; ++it) {
        const int m0 = it * 128;
        {   // prefetch next m-tile
            const int mn = m0 + 128 + (tid >> 1);
            if (mn < M_SP) {
                __builtin_prefetch(KpT + ((size_t)(b * M_SP + mn)) * 128, 0, 3);
                __builtin_prefetch(Gp + ((size_t)(b * 128 + (tid >> 1))) * M_SP + m0 + 128, 0, 3);
            }
        }
        __syncthreads();   // previous iter's K/G reads done
        if (m0 + 128 <= M_SP) {
            {   // Ks[mm][c] = KpT[b][m0+mm][c]
                const int mm = tid >> 1, seg = (tid & 1) * 64;
                const u16* src = KpT + ((size_t)(b * M_SP + m0 + mm)) * 128 + seg;
                u16* dst = Ks + mm * 128 + seg;
#pragma unroll
                for (int j = 0; j < 8; ++j) cp_b128(src + 8 * j, dst + 8 * j);
            }
            {   // Gs[c][mm] = Gp[b][c][m0+mm]
                const int c = tid >> 1, seg = (tid & 1) * 64;
                const u16* src = Gp + ((size_t)(b * 128 + c)) * M_SP + m0 + seg;
                u16* dst = Gs + c * 128 + seg;
#pragma unroll
                for (int j = 0; j < 8; ++j) cp_b128(src + 8 * j, dst + 8 * j);
            }
            cp_wait();
        } else {   // masked tail tile: scalar guarded copies, zero fill
            {
                const int mm = tid >> 1, seg = (tid & 1) * 64;
                const int m = m0 + mm;
                uint4* dst = (uint4*)(Ks + mm * 128 + seg);
                if (m < M_SP) {
                    const uint4* src = (const uint4*)(KpT + ((size_t)(b * M_SP + m)) * 128 + seg);
#pragma unroll
                    for (int j = 0; j < 8; ++j) dst[j] = src[j];
                } else {
                    const uint4 z = {0u, 0u, 0u, 0u};
#pragma unroll
                    for (int j = 0; j < 8; ++j) dst[j] = z;
                }
            }
            {
                const int c = tid >> 1, seg = (tid & 1) * 64;
                for (int j = 0; j < 64; ++j) {
                    const int m = m0 + seg + j;
                    Gs[c * 128 + seg + j] =
                        (m < M_SP) ? Gp[((size_t)(b * 128 + c)) * M_SP + m] : (u16)0;
                }
            }
        }
        __syncthreads();

        // S = Q x K^T   (K dim = Ci = 128 -> 4 wmma steps)
        v8f S[8];
#pragma unroll
        for (int t = 0; t < 8; ++t) S[t] = zero_v8();
#pragma unroll
        for (int kk = 0; kk < 4; ++kk) {
            v16bf bk[8];
#pragma unroll
            for (int t = 0; t < 8; ++t) bk[t] = load_frag16(Ks, 128, t * 16, kk * 32);
#pragma unroll
            for (int t = 0; t < 8; ++t) S[t] = wmma_bf16(qf[kk], bk[t], S[t]);
        }
        // mask columns past M
#pragma unroll
        for (int t = 0; t < 8; ++t) {
            if (m0 + t * 16 + col >= M_SP) {
#pragma unroll
                for (int r = 0; r < 8; ++r) S[t][r] = -3.0e38f;
            }
        }
        // online softmax (row = C-layout: vgpr r, lane-half selects +8)
        float nmax[8], resc[8];
#pragma unroll
        for (int r = 0; r < 8; ++r) {
            float tm = S[0][r];
#pragma unroll
            for (int t = 1; t < 8; ++t) tm = fmaxf(tm, S[t][r]);
            for (int d = 1; d < 16; d <<= 1) tm = fmaxf(tm, __shfl_xor(tm, d, 32));
            nmax[r] = fmaxf(rmax[r], tm);
            resc[r] = __expf(rmax[r] - nmax[r]);
            rmax[r] = nmax[r];
        }
        float ps[8];
#pragma unroll
        for (int r = 0; r < 8; ++r) ps[r] = 0.f;
#pragma unroll
        for (int t = 0; t < 8; ++t) {
#pragma unroll
            for (int r = 0; r < 8; ++r) {
                const float p = __expf(S[t][r] - nmax[r]);
                ps[r] += p;
                Pw[(r + hi8) * 128 + t * 16 + col] = f2bf(p);
            }
        }
#pragma unroll
        for (int r = 0; r < 8; ++r) {
            float s = ps[r];
            for (int d = 1; d < 16; d <<= 1) s += __shfl_xor(s, d, 32);
            rsum[r] = rsum[r] * resc[r] + s;
        }
#pragma unroll
        for (int t = 0; t < 8; ++t)
#pragma unroll
            for (int r = 0; r < 8; ++r) O[t][r] *= resc[r];
        __syncthreads();   // P strips visible to fragment loads

        // O += P x G^T   (K dim = m tile = 128 -> 4 wmma steps)
#pragma unroll
        for (int kk = 0; kk < 4; ++kk) {
            const v16bf a = load_frag16(Pw, 128, 0, kk * 32);
            v16bf bg[8];
#pragma unroll
            for (int t = 0; t < 8; ++t) bg[t] = load_frag16(Gs, 128, t * 16, kk * 32);
#pragma unroll
            for (int t = 0; t < 8; ++t) O[t] = wmma_bf16(a, bg[t], O[t]);
        }
    }
    // normalize and write Y[b][n][c]
#pragma unroll
    for (int t = 0; t < 8; ++t) {
#pragma unroll
        for (int r = 0; r < 8; ++r) {
            const float v = O[t][r] / rsum[r];
            const int n = n0 + w * 16 + r + hi8;
            Y[((size_t)(b * N_SP + n)) * 128 + t * 16 + col] = f2bf(v);
        }
    }
}

// ---------------------------------------------------------------------------
// Kernel 6: out = relu(Ww y + biasW + x), fp32 out. K=128 in 4 steps.
// ---------------------------------------------------------------------------
__global__ void __launch_bounds__(256) k_out(const u16* __restrict__ Ww,
                                             const float* __restrict__ biasW,
                                             const u16* __restrict__ Y,
                                             const float* __restrict__ x,
                                             float* __restrict__ out) {
    __shared__ __align__(16) u16 As[128 * 32];
    __shared__ __align__(16) u16 Bs[128 * 32];
    const int n0  = blockIdx.x * 128;
    const int r0  = blockIdx.y * 128;
    const int b   = blockIdx.z;
    const int tid = threadIdx.x;
    const int w = tid >> 5, lid = tid & 31;
    const int col = lid & 15, hi8 = (lid >> 4) << 3;

    v8f acc[8];
#pragma unroll
    for (int t = 0; t < 8; ++t) acc[t] = zero_v8();

    for (int ks = 0; ks < 4; ++ks) {
        if (ks + 1 < 4) {
            __builtin_prefetch(Ww + (r0 + (tid >> 1)) * 128 + (ks + 1) * 32, 0, 3);
            __builtin_prefetch(Y + ((size_t)(b * N_SP + n0 + (tid >> 1))) * 128 + (ks + 1) * 32, 0, 3);
        }
        __syncthreads();
        {
            const int row = tid >> 1, seg = (tid & 1) * 16;
            const u16* src = Ww + (r0 + row) * 128 + ks * 32 + seg;
            u16* dst = As + row * 32 + seg;
            cp_b128(src, dst);
            cp_b128(src + 8, dst + 8);
        }
        {   // Y is [n][ci] already -> direct copy, no transpose
            const int n = tid >> 1, seg = (tid & 1) * 16;
            const u16* src = Y + ((size_t)(b * N_SP + n0 + n)) * 128 + ks * 32 + seg;
            u16* dst = Bs + n * 32 + seg;
            cp_b128(src, dst);
            cp_b128(src + 8, dst + 8);
        }
        cp_wait();
        __syncthreads();
        const v16bf a = load_frag16(As, 32, w * 16, 0);
        v16bf bfr[8];
#pragma unroll
        for (int t = 0; t < 8; ++t) bfr[t] = load_frag16(Bs, 32, t * 16, 0);
#pragma unroll
        for (int t = 0; t < 8; ++t) acc[t] = wmma_bf16(a, bfr[t], acc[t]);
    }
#pragma unroll
    for (int t = 0; t < 8; ++t) {
#pragma unroll
        for (int r = 0; r < 8; ++r) {
            const int row = r0 + w * 16 + r + hi8;
            const int n = n0 + t * 16 + col;
            const size_t gi = ((size_t)(b * 256 + row)) * N_SP + n;
            const float v = acc[t][r] + biasW[row] + x[gi];
            out[gi] = fmaxf(v, 0.f);
        }
    }
}

// ---------------------------------------------------------------------------
extern "C" void kernel_launch(void* const* d_in, const int* in_sizes, int n_in,
                              void* d_out, int out_size, void* d_ws, size_t ws_size,
                              hipStream_t stream) {
    const float* x       = (const float*)d_in[0];
    const float* w_theta = (const float*)d_in[1];
    const float* b_theta = (const float*)d_in[2];
    const float* w_phi   = (const float*)d_in[3];
    const float* b_phi   = (const float*)d_in[4];
    const float* w_g     = (const float*)d_in[5];
    const float* b_g     = (const float*)d_in[6];
    const float* g1      = (const float*)d_in[7];
    const float* be1     = (const float*)d_in[8];
    const float* mu1     = (const float*)d_in[9];
    const float* va1     = (const float*)d_in[10];
    const float* w_W     = (const float*)d_in[11];
    const float* b_W     = (const float*)d_in[12];
    const float* g2      = (const float*)d_in[13];
    const float* be2     = (const float*)d_in[14];
    const float* mu2     = (const float*)d_in[15];
    const float* va2     = (const float*)d_in[16];
    (void)in_sizes; (void)n_in; (void)out_size; (void)ws_size;

    char* ws = (char*)d_ws;
    size_t off = 0;
    auto carve = [&](size_t bytes) -> void* {
        void* p = ws + off;
        off += (bytes + 255) & ~((size_t)255);
        return p;
    };
    u16*   Xbf   = (u16*)  carve((size_t)B_N * C_IN * N_SP * 2);
    u16*   Wcat  = (u16*)  carve((size_t)384 * 256 * 2);
    float* bias1 = (float*)carve((size_t)384 * 4);
    u16*   Ww    = (u16*)  carve((size_t)256 * 128 * 2);
    float* biasW = (float*)carve((size_t)256 * 4);
    u16*   P     = (u16*)  carve((size_t)B_N * 384 * N_SP * 2);
    u16*   KpT   = (u16*)  carve((size_t)B_N * M_SP * 128 * 2);
    u16*   Gp    = (u16*)  carve((size_t)B_N * 128 * M_SP * 2);
    u16*   Yb    = (u16*)  carve((size_t)B_N * N_SP * 128 * 2);

    k_fold<<<640, 256, 0, stream>>>(w_theta, b_theta, w_phi, b_phi, w_g, b_g,
                                    g1, be1, mu1, va1, w_W, b_W, g2, be2, mu2, va2,
                                    Wcat, bias1, Ww, biasW);
    const int nx = B_N * C_IN * N_SP;
    k_cvt<<<(nx + 255) / 256, 256, 0, stream>>>(x, Xbf, nx);
    k_proj<<<dim3(N_SP / 128, 3, B_N), 256, 0, stream>>>(Wcat, bias1, Xbf, P);
    const int np = B_N * C_I * M_SP;
    k_pool<<<(np + 255) / 256, 256, 0, stream>>>(P, KpT, Gp);
    k_attn<<<dim3(N_SP / 128, B_N), 256, 0, stream>>>(P, KpT, Gp, Yb);
    k_out<<<dim3(N_SP / 128, 2, B_N), 256, 0, stream>>>(Ww, biasW, Yb, x, (float*)d_out);
}
